// PerformerClassifier_61435212202450
// MI455X (gfx1250) — matrix operand order repaired and checked by reading, not compiled, not run on hardware
//
#include <hip/hip_runtime.h>

// ---------------------------------------------------------------------------
// Performer classifier fwd: V=32000 D=512 C=2 S=2048 L=4 M=64 B=16, H=2048
// bf16 WMMA GEMMs fed by GLOBAL_LOAD_ASYNC_TO_LDS_B128 (double-buffered LDS)
// ---------------------------------------------------------------------------
#define SEQ    2048
#define DM     512
#define HD     2048
#define BATCH  16
#define NTOK   (BATCH*SEQ)   // 32768
#define MF     64
#define NLAYER 4
#define CHROWS 4096
#define NCHUNK (NTOK/CHROWS)

enum { ACT_NONE = 0, ACT_GELU = 1, ACT_ELU1 = 2 };

typedef __attribute__((ext_vector_type(4)))  float        f32x4;
typedef __attribute__((ext_vector_type(4)))  unsigned int u32x4;
typedef __attribute__((ext_vector_type(2)))  unsigned int u32x2;
typedef __attribute__((ext_vector_type(16))) __bf16       v16bf;
typedef __attribute__((ext_vector_type(8)))  float        v8f;

struct Frag32B { u32x4 lo, hi; };   // 32B -> v16bf

__device__ __forceinline__ unsigned short f2bf(float f) {
  unsigned u = __float_as_uint(f);
  u += 0x7FFFu + ((u >> 16) & 1u);   // RNE
  return (unsigned short)(u >> 16);
}

// Async DMA: 16 bytes global -> LDS, tracked by ASYNCcnt (CDNA5).
// GVS mode: mem = SGPR64 base + VGPR32 offset; LDS dest from VGPR (per-lane,
// so padded LDS rows work). Generic LDS pointers carry the LDS byte offset in
// the low 32 bits (ISA 10.2 aperture rules), so truncation yields VDST addr.
__device__ __forceinline__ void async_b128(unsigned lds, unsigned goff,
                                           unsigned long long base) {
  asm volatile("global_load_async_to_lds_b128 %0, %1, %2"
               :: "v"(lds), "v"(goff), "s"(base) : "memory");
}
__device__ __forceinline__ void wait_async() {
  asm volatile("s_wait_asynccnt 0x0" ::: "memory");
}

// ---------------------------------------------------------------------------
// Yf/Yh = act( rowscale * (X @ Wt^T) + bias ) + res
//   X  : bf16 [rows][K]   (row-major)
//   Wt : bf16 [N2][K]     (weights pre-transposed+converted once per launch)
// Block tile 128x128, BK=32 = one v_wmma_f32_16x16x32_bf16 step.
// 8 waves in 4x2 grid, each owns 32x64 (2x4 frags). Double-buffered LDS fed
// by async-to-LDS; frag loads are plain ds_load_b128 per the ISA striping:
//   A lane(row=l16): K = hf*8+[0..7] , 16+hf*8+[0..7]
//   B lane(col=l16): K = hf*8+[0..7] , 16+hf*8+[0..7]
// ---------------------------------------------------------------------------
#define BM   128
#define BN   128
#define BKK  32
#define ASTR 40   // padded row stride (elems): 80B keeps 16B-aligned chunks
#define BSTR 40

__global__ __launch_bounds__(256)
void wmma_gemm_kernel(const unsigned short* __restrict__ Xb,
                      const unsigned short* __restrict__ Wtb,
                      const float* __restrict__ bias,
                      const float* __restrict__ rowscale,
                      const float* __restrict__ resb,
                      float* __restrict__ Yfb,
                      unsigned short* __restrict__ Yhb,
                      int K, int N2, int act,
                      long long sxb, long long swb, long long syb, long long srs)
{
  __shared__ unsigned short As[2][BM * ASTR];
  __shared__ unsigned short Bs[2][BN * BSTR];

  const int tid  = threadIdx.x;
  const int lane = tid & 31, wid = tid >> 5;
  const int wm = wid & 3, wn = wid >> 2;
  const int l16 = lane & 15, hf = lane >> 4;
  const int row0 = blockIdx.x * BM;
  const int n0   = blockIdx.y * BN;
  const long long zbv = blockIdx.z;

  const unsigned short* X  = Xb  + zbv * sxb;
  const unsigned short* Wt = Wtb + zbv * swb;
  const unsigned long long xbase = (unsigned long long)(X  + (long long)row0 * K);
  const unsigned long long wbase = (unsigned long long)(Wt + (long long)n0   * K);

  const unsigned ldsA[2] = { (unsigned)(unsigned long long)&As[0][0],
                             (unsigned)(unsigned long long)&As[1][0] };
  const unsigned ldsB[2] = { (unsigned)(unsigned long long)&Bs[0][0],
                             (unsigned)(unsigned long long)&Bs[1][0] };

  // 512 16B-chunks per tile (128 rows x 4), 2 per thread.
  const int c0 = tid, c1 = tid + 256;
  const int r0c = c0 >> 2, o0c = (c0 & 3) << 3;   // row, elem-offset(8 bf16)
  const int r1c = c1 >> 2, o1c = (c1 & 3) << 3;

  auto fill = [&](int k0, int bsel) {
    async_b128(ldsA[bsel] + (unsigned)((r0c * ASTR + o0c) * 2),
               (unsigned)((r0c * K + k0 + o0c) * 2), xbase);
    async_b128(ldsA[bsel] + (unsigned)((r1c * ASTR + o1c) * 2),
               (unsigned)((r1c * K + k0 + o1c) * 2), xbase);
    if (n0 + r0c < N2)                    // cols >= N2: leave stale, masked on store
      async_b128(ldsB[bsel] + (unsigned)((r0c * BSTR + o0c) * 2),
                 (unsigned)((r0c * K + k0 + o0c) * 2), wbase);
    if (n0 + r1c < N2)
      async_b128(ldsB[bsel] + (unsigned)((r1c * BSTR + o1c) * 2),
                 (unsigned)((r1c * K + k0 + o1c) * 2), wbase);
  };

  v8f acc[2][4];
#pragma unroll
  for (int mi = 0; mi < 2; mi++)
#pragma unroll
    for (int ni = 0; ni < 4; ni++)
#pragma unroll
      for (int r = 0; r < 8; r++) acc[mi][ni][r] = 0.0f;

  fill(0, 0);
  wait_async();
  __syncthreads();

  const int nk = K / BKK;
  for (int i = 0; i < nk; i++) {
    const int bsel = i & 1;
    if (i + 1 < nk) fill((i + 1) * BKK, bsel ^ 1);   // DMA overlaps WMMA below

    v16bf a[2], b[4];
#pragma unroll
    for (int mi = 0; mi < 2; mi++) {
      const unsigned short* ap = &As[bsel][(wm * 32 + mi * 16 + l16) * ASTR];
      Frag32B fr;
      fr.lo = *(const u32x4*)&ap[hf * 8];
      fr.hi = *(const u32x4*)&ap[16 + hf * 8];
      a[mi] = __builtin_bit_cast(v16bf, fr);
    }
#pragma unroll
    for (int ni = 0; ni < 4; ni++) {
      const unsigned short* bp = &Bs[bsel][(wn * 64 + ni * 16 + l16) * BSTR];
      Frag32B fr;
      fr.lo = *(const u32x4*)&bp[hf * 8];
      fr.hi = *(const u32x4*)&bp[16 + hf * 8];
      b[ni] = __builtin_bit_cast(v16bf, fr);
    }
#pragma unroll
    for (int mi = 0; mi < 2; mi++)
#pragma unroll
      for (int ni = 0; ni < 4; ni++)
        acc[mi][ni] = __builtin_amdgcn_wmma_f32_16x16x32_bf16(
            false, a[mi], false, b[ni], (short)0, acc[mi][ni], false, false);

    wait_async();      // next tile resident (this wave's DMA)
    __syncthreads();   // ... and everyone's; also fences LDS reuse
  }

  // ---- epilogue: rowscale, bias, act, residual; dual f32/bf16 stores ----
  const float*    res = resb ? resb + zbv * syb : nullptr;
  float*          Yf  = Yfb  ? Yfb  + zbv * syb : nullptr;
  unsigned short* Yh  = Yhb  ? Yhb  + zbv * syb : nullptr;
  const float*    rs  = rowscale ? rowscale + zbv * srs : nullptr;
#pragma unroll
  for (int mi = 0; mi < 2; mi++) {
#pragma unroll
    for (int ni = 0; ni < 4; ni++) {
      int gcol = n0 + wn * 64 + ni * 16 + l16;
      if (gcol < N2) {
        float bb = bias ? bias[gcol] : 0.0f;
#pragma unroll
        for (int r = 0; r < 8; r++) {
          int grow = row0 + wm * 32 + mi * 16 + hf * 8 + r;   // C: M = r + hf*8
          float val = acc[mi][ni][r];
          if (rs) val *= rs[grow];
          val += bb;
          if (act == ACT_GELU) {
            val = 0.5f * val * (1.0f + erff(val * 0.70710678118654752f));
          } else if (act == ACT_ELU1) {
            val = (val > 0.0f) ? (val + 1.0f) : __expf(val);
          }
          long long o = (long long)grow * N2 + gcol;
          if (res) val += res[o];
          if (Yf) Yf[o] = val;
          if (Yh) Yh[o] = f2bf(val);
        }
      }
    }
  }
}

// ---------------------------------------------------------------------------
// Weight convert+transpose (once per launch): src f32 [z][K][N] -> dst bf16 [z][N][K]
__global__ __launch_bounds__(256)
void convT_kernel(const float* __restrict__ src, unsigned short* __restrict__ dst,
                  int K, int N)
{
  long long mb = (long long)blockIdx.z * K * N;
  int idx = blockIdx.x * 256 + threadIdx.x;
  if (idx >= K * N) return;
  int k = idx / N, n = idx - k * N;
  dst[mb + (long long)n * K + k] = f2bf(src[mb + idx]);
}

// h = embed[x] + pos : write f32 + bf16
__global__ __launch_bounds__(256)
void embed_kernel(const int* __restrict__ x, const float* __restrict__ emb,
                  const float* __restrict__ pos, float* __restrict__ hf32,
                  unsigned short* __restrict__ hbf)
{
  long long idx = (long long)blockIdx.x * 256 + threadIdx.x;
  int n  = (int)(idx >> 7);
  int dj = ((int)idx & 127) << 2;
  int tok = x[n];
  int s = n & (SEQ - 1);
  f32x4 e = *(const f32x4*)&emb[(long long)tok * DM + dj];
  f32x4 p = *(const f32x4*)&pos[(long long)s * DM + dj];
  f32x4 o = {e.x + p.x, e.y + p.y, e.z + p.z, e.w + p.w};
  *(f32x4*)&hf32[(long long)n * DM + dj] = o;
  u32x2 pk2;
  pk2.x = (unsigned)f2bf(o.x) | ((unsigned)f2bf(o.y) << 16);
  pk2.y = (unsigned)f2bf(o.z) | ((unsigned)f2bf(o.w) << 16);
  *(u32x2*)&hbf[(long long)n * DM + dj] = pk2;
}

// kvT[b][d][m] = bf16( sum_s pk[b,s,m]*v[b,s,d] );  pksum[b][m] = sum_s pk
__global__ __launch_bounds__(256)
void kv_kernel(const float* __restrict__ pk, const float* __restrict__ v,
               unsigned short* __restrict__ kvT, float* __restrict__ pksum)
{
  const int b  = blockIdx.y;
  const int d0 = blockIdx.x * 64;
  const int tid = threadIdx.x;
  const int m = tid & 63;
  const int g = tid >> 6;
  __shared__ float sp[64];
  __shared__ float sv[64];
  float acc[16];
#pragma unroll
  for (int i = 0; i < 16; i++) acc[i] = 0.f;
  float psum = 0.f;
  const long long pkbase = (long long)b * SEQ * MF;
  const long long vbase  = (long long)b * SEQ * DM;
  for (int s = 0; s < SEQ; s++) {
    if (tid < 64)       sp[tid]      = pk[pkbase + (long long)s * MF + tid];
    else if (tid < 128) sv[tid - 64] = v[vbase + (long long)s * DM + d0 + (tid - 64)];
    __syncthreads();
    float p = sp[m];
    if (g == 0) psum += p;
#pragma unroll
    for (int i = 0; i < 16; i++) acc[i] += p * sv[g * 16 + i];
    __syncthreads();
  }
  unsigned short* o = kvT + (long long)b * DM * MF;
#pragma unroll
  for (int i = 0; i < 16; i++) o[(long long)(d0 + g * 16 + i) * MF + m] = f2bf(acc[i]);
  if (g == 0 && blockIdx.x == 0) pksum[b * MF + m] = psum;
}

// zinv[n] = 1 / (dot(pq[n,:], pksum[b,:]) + 1e-6)
__global__ __launch_bounds__(256)
void zinv_kernel(const float* __restrict__ pq, const float* __restrict__ pksum,
                 float* __restrict__ zinv)
{
  int n = blockIdx.x * 256 + threadIdx.x;
  int b = n >> 11;
  const float* pr = pq + (long long)n * MF;
  const float* ps = pksum + b * MF;
  float acc = 0.f;
#pragma unroll
  for (int mm = 0; mm < MF; mm++) acc += pr[mm] * ps[mm];
  zinv[n] = 1.0f / (acc + 1e-6f);
}

// LayerNorm(D=512) -> bf16, one row per block
__global__ __launch_bounds__(256)
void ln_kernel(const float* __restrict__ x, const float* __restrict__ g,
               const float* __restrict__ bta, unsigned short* __restrict__ y)
{
  __shared__ float red[256];
  const int n = blockIdx.x;
  const int t = threadIdx.x;
  const float* xr = x + (long long)n * DM;
  float v0 = xr[t], v1 = xr[t + 256];
  red[t] = v0 + v1;
  __syncthreads();
  for (int st = 128; st > 0; st >>= 1) { if (t < st) red[t] += red[t + st]; __syncthreads(); }
  float mu = red[0] * (1.0f / DM);
  __syncthreads();
  float d0 = v0 - mu, d1 = v1 - mu;
  red[t] = d0 * d0 + d1 * d1;
  __syncthreads();
  for (int st = 128; st > 0; st >>= 1) { if (t < st) red[t] += red[t + st]; __syncthreads(); }
  float inv = rsqrtf(red[0] * (1.0f / DM) + 1e-5f);
  unsigned short* yr = y + (long long)n * DM;
  yr[t]       = f2bf(d0 * inv * g[t]       + bta[t]);
  yr[t + 256] = f2bf(d1 * inv * g[t + 256] + bta[t + 256]);
}

// classifier head on h[:,0,:] (f32)
__global__ __launch_bounds__(256)
void classifier_kernel(const float* __restrict__ h, const float* __restrict__ Wc1,
                       const float* __restrict__ bc1, const float* __restrict__ Wc2,
                       const float* __restrict__ bc2, float* __restrict__ out)
{
  __shared__ float sp[BATCH * DM];
  __shared__ float sh[BATCH * (DM / 2)];
  const int t = threadIdx.x;
  for (int i = t; i < BATCH * DM; i += 256) {
    int b = i / DM, d = i % DM;
    sp[i] = h[(long long)b * SEQ * DM + d];
  }
  __syncthreads();
  for (int j = t; j < BATCH * (DM / 2); j += 256) {
    int b = j / (DM / 2), o = j % (DM / 2);
    float acc = bc1[o];
    for (int d = 0; d < DM; d++) acc += sp[b * DM + d] * Wc1[d * (DM / 2) + o];
    sh[j] = acc > 0.f ? acc : 0.f;
  }
  __syncthreads();
  for (int j = t; j < BATCH * 2; j += 256) {
    int b = j >> 1, c = j & 1;
    float acc = bc2[c];
    for (int o = 0; o < DM / 2; o++) acc += sh[b * (DM / 2) + o] * Wc2[o * 2 + c];
    out[j] = acc;
  }
}

// ---------------------------------------------------------------------------
static inline void gemm(hipStream_t st, const unsigned short* X, const unsigned short* Wt,
                        const float* bias, const float* rowscale, const float* res,
                        float* Yf, unsigned short* Yh, int rows, int K, int N2, int act,
                        int batches = 1, long long sxb = 0, long long swb = 0,
                        long long syb = 0, long long srs = 0)
{
  dim3 grid(rows / BM, (N2 + BN - 1) / BN, batches);
  wmma_gemm_kernel<<<grid, 256, 0, st>>>(X, Wt, bias, rowscale, res, Yf, Yh,
                                         K, N2, act, sxb, swb, syb, srs);
}

extern "C" void kernel_launch(void* const* d_in, const int* in_sizes, int n_in,
                              void* d_out, int out_size, void* d_ws, size_t ws_size,
                              hipStream_t stream)
{
  const int*   x     = (const int*)  d_in[0];
  const float* embed = (const float*)d_in[1];
  const float* pos   = (const float*)d_in[2];
  const float* Wq    = (const float*)d_in[3];
  const float* bq    = (const float*)d_in[4];
  const float* Wk    = (const float*)d_in[5];
  const float* bk    = (const float*)d_in[6];
  const float* Wv    = (const float*)d_in[7];
  const float* bv    = (const float*)d_in[8];
  const float* P     = (const float*)d_in[9];
  const float* Wo    = (const float*)d_in[10];
  const float* bo    = (const float*)d_in[11];
  const float* ln_g  = (const float*)d_in[12];
  const float* ln_b  = (const float*)d_in[13];
  const float* W1    = (const float*)d_in[14];
  const float* b1    = (const float*)d_in[15];
  const float* W2    = (const float*)d_in[16];
  const float* b2    = (const float*)d_in[17];
  const float* Wc1   = (const float*)d_in[18];
  const float* bc1   = (const float*)d_in[19];
  const float* Wc2   = (const float*)d_in[20];
  const float* bc2   = (const float*)d_in[21];
  (void)in_sizes; (void)n_in; (void)out_size; (void)ws_size;

  const long long ND = (long long)NTOK * DM;
  char* wp = (char*)d_ws;
  auto alloc = [&](size_t bytes) -> char* {
    char* p = wp; wp += (bytes + 255) & ~(size_t)255; return p;
  };
  float* h_a   = (float*)alloc(ND * 4);
  float* h_b   = (float*)alloc(ND * 4);
  float* v_f   = (float*)alloc(ND * 4);                 // v / att@Wo
  float* pqf   = (float*)alloc((long long)NTOK * MF * 4);
  float* pkf   = (float*)alloc((long long)NTOK * MF * 4);
  float* psumb = (float*)alloc(BATCH * MF * 4);
  float* zinv  = (float*)alloc(NTOK * 4);
  unsigned short* h_bf   = (unsigned short*)alloc(ND * 2);
  unsigned short* qk_bf  = (unsigned short*)alloc(ND * 2);   // q/k/att/hn
  unsigned short* pq_bf  = (unsigned short*)alloc((long long)NTOK * MF * 2);
  unsigned short* kvT_bf = (unsigned short*)alloc((long long)BATCH * MF * DM * 2);
  unsigned short* ffn_bf = (unsigned short*)alloc((long long)CHROWS * HD * 2);
  unsigned short* WqT = (unsigned short*)alloc((long long)NLAYER * DM * DM * 2);
  unsigned short* WkT = (unsigned short*)alloc((long long)NLAYER * DM * DM * 2);
  unsigned short* WvT = (unsigned short*)alloc((long long)NLAYER * DM * DM * 2);
  unsigned short* WoT = (unsigned short*)alloc((long long)NLAYER * DM * DM * 2);
  unsigned short* PT  = (unsigned short*)alloc((long long)NLAYER * DM * MF * 2);
  unsigned short* W1T = (unsigned short*)alloc((long long)NLAYER * DM * HD * 2);
  unsigned short* W2T = (unsigned short*)alloc((long long)NLAYER * HD * DM * 2);

  // weights -> transposed bf16 (deterministic, every launch)
  convT_kernel<<<dim3((DM*DM + 255)/256, 1, NLAYER), 256, 0, stream>>>(Wq, WqT, DM, DM);
  convT_kernel<<<dim3((DM*DM + 255)/256, 1, NLAYER), 256, 0, stream>>>(Wk, WkT, DM, DM);
  convT_kernel<<<dim3((DM*DM + 255)/256, 1, NLAYER), 256, 0, stream>>>(Wv, WvT, DM, DM);
  convT_kernel<<<dim3((DM*DM + 255)/256, 1, NLAYER), 256, 0, stream>>>(Wo, WoT, DM, DM);
  convT_kernel<<<dim3((DM*MF + 255)/256, 1, NLAYER), 256, 0, stream>>>(P,  PT,  DM, MF);
  convT_kernel<<<dim3((DM*HD + 255)/256, 1, NLAYER), 256, 0, stream>>>(W1, W1T, DM, HD);
  convT_kernel<<<dim3((HD*DM + 255)/256, 1, NLAYER), 256, 0, stream>>>(W2, W2T, HD, DM);

  embed_kernel<<<(NTOK * (DM/4)) / 256, 256, 0, stream>>>(x, embed, pos, h_a, h_bf);

  float* hc = h_a;
  float* hn = h_b;
  for (int l = 0; l < NLAYER; l++) {
    const unsigned short* WqT_l = WqT + (long long)l * DM * DM;
    const unsigned short* WkT_l = WkT + (long long)l * DM * DM;
    const unsigned short* WvT_l = WvT + (long long)l * DM * DM;
    const unsigned short* WoT_l = WoT + (long long)l * DM * DM;
    const unsigned short* PT_l  = PT  + (long long)l * DM * MF;
    const unsigned short* W1T_l = W1T + (long long)l * DM * HD;
    const unsigned short* W2T_l = W2T + (long long)l * HD * DM;
    const float* bq_l = bq + l * DM;   const float* bk_l = bk + l * DM;
    const float* bv_l = bv + l * DM;   const float* bo_l = bo + l * DM;
    const float* b1_l = b1 + l * HD;   const float* b2_l = b2 + l * DM;
    const float* lg_l = ln_g + l * DM; const float* lb_l = ln_b + l * DM;

    // q -> pq (reuse qk_bf), then k -> pk, then v
    gemm(stream, h_bf, WqT_l, bq_l, nullptr, nullptr, nullptr, qk_bf, NTOK, DM, DM, ACT_NONE);
    gemm(stream, qk_bf, PT_l, nullptr, nullptr, nullptr, pqf, pq_bf, NTOK, DM, MF, ACT_ELU1);
    gemm(stream, h_bf, WkT_l, bk_l, nullptr, nullptr, nullptr, qk_bf, NTOK, DM, DM, ACT_NONE);
    gemm(stream, qk_bf, PT_l, nullptr, nullptr, nullptr, pkf, nullptr, NTOK, DM, MF, ACT_ELU1);
    gemm(stream, h_bf, WvT_l, bv_l, nullptr, nullptr, v_f, nullptr, NTOK, DM, DM, ACT_NONE);

    kv_kernel<<<dim3(DM/64, BATCH), 256, 0, stream>>>(pkf, v_f, kvT_bf, psumb);
    zinv_kernel<<<NTOK/256, 256, 0, stream>>>(pqf, psumb, zinv);

    // att = (pq @ kv) * zinv  -> bf16 (batched; kvT is the [N][K] operand)
    gemm(stream, pq_bf, kvT_bf, nullptr, zinv, nullptr, nullptr, qk_bf,
         SEQ, MF, DM, ACT_NONE, BATCH,
         (long long)SEQ * MF, (long long)MF * DM, (long long)SEQ * DM, (long long)SEQ);

    // att @ Wo + bo -> f32 ; layernorm -> bf16 hn (reuse qk_bf)
    gemm(stream, qk_bf, WoT_l, bo_l, nullptr, nullptr, v_f, nullptr, NTOK, DM, DM, ACT_NONE);
    ln_kernel<<<NTOK, 256, 0, stream>>>(v_f, lg_l, lb_l, qk_bf);

    // FFN chunks: h_next = gelu(hn@W1+b1)@W2 + b2 + h_cur
    for (int c = 0; c < NCHUNK; c++) {
      long long off = (long long)c * CHROWS * DM;
      gemm(stream, qk_bf + off, W1T_l, b1_l, nullptr, nullptr, nullptr, ffn_bf,
           CHROWS, DM, HD, ACT_GELU);
      gemm(stream, ffn_bf, W2T_l, b2_l, nullptr, hc + off, hn + off, h_bf + off,
           CHROWS, HD, DM, ACT_NONE);
    }

    float* tmp = hc; hc = hn; hn = tmp;
  }

  classifier_kernel<<<1, 256, 0, stream>>>(hc, Wc1, bc1, Wc2, bc2, (float*)d_out);
}
